// timeResidualHyperGNNs_74955769250644
// MI455X (gfx1250) — matrix operand order repaired and checked by reading, not compiled, not run on hardware
//
#include <hip/hip_runtime.h>
#include <hip/hip_bf16.h>
#include <math.h>

#define B_    64
#define R_    400
#define N_    25600
#define E_    25600
#define NNZ_  256000
#define HC_   256
#define L_    3
#define HID_  512
#define NC_   2
#define CV_   80200
#define IN1_  (CV_ + HC_ * L_)   /* 80968 */
#define EPS_  1e-5f

typedef __attribute__((ext_vector_type(16))) __bf16 v16bf;
typedef __attribute__((ext_vector_type(8)))  float  v8f;

__device__ __forceinline__ unsigned short f2bf(float f) {
    union { float f; unsigned int u; } x; x.f = f;
    unsigned int r = (x.u + 0x7FFFu + ((x.u >> 16) & 1u)) >> 16;
    return (unsigned short)r;
}

__device__ __forceinline__ int triu_off(int r) {
    return r * R_ - (r * (r - 1)) / 2;
}

// ---------------------------------------------------------------------------
// WMMA bf16 GEMM: C[M,Nc] = A[M,K](f32) x B[K,Nc](f32) (+bias[n]), f32 accum.
// Block: 256 threads = 8 waves; tile 128(M) x 128(N); K-step 32.
// Wave grid 4(M) x 2(N); each wave: 2x4 WMMA 16x16 tiles = 8 wmma / K-step.
// Staging is branchless: clamped addresses + vector selects (K, lda, ldb, Nc
// are all multiples of 4, so float4 K-groups never straddle the K edge).
// ---------------------------------------------------------------------------
#define TM 128
#define TN 128
#define TK 32
#define AST 40   /* LDS row stride in halfwords: 80B, 16B aligned, bank-spread */
#define BST 40

__global__ __launch_bounds__(256)
void k_gemm(const float* __restrict__ A, const float* __restrict__ Bm,
            const float* __restrict__ bias, float* __restrict__ C,
            int M, int Nc, int K, int lda, int ldb, int ldc)
{
    __shared__ __align__(16) unsigned short As[TM * AST];
    __shared__ __align__(16) unsigned short Bs[TN * BST];

    const int t     = threadIdx.x;
    const int lane  = t & 31;
    const int wave  = t >> 5;
    const int waveM = wave & 3;         // 4 waves along M (32 rows each)
    const int waveN = wave >> 2;        // 2 waves along N (64 cols each)
    const int half  = lane >> 4;
    const int l16   = lane & 15;
    const int blockN = blockIdx.x * TN;
    const int blockM = blockIdx.y * TM;

    // Per-thread staging coordinates (loop-invariant)
    const int a_r  = t >> 3;            // 8 float4 per 32-wide A row
    const int a_kq = (t & 7) << 2;
    const int b_k  = t >> 5;            // 32 float4 per 128-wide B row
    const int b_nq = (t & 31) << 2;

    v8f acc[2][4];
    #pragma unroll
    for (int sm = 0; sm < 2; ++sm)
        #pragma unroll
        for (int sn = 0; sn < 4; ++sn)
            acc[sm][sn] = (v8f){};

    const int ksteps = (K + TK - 1) / TK;
    for (int ks = 0; ks < ksteps; ++ks) {
        const int k0 = ks * TK;

        // ---- stage A tile: 4x (float4 load -> packed bf16 ds_store_b64) ----
        #pragma unroll
        for (int i = 0; i < 4; ++i) {
            int r  = a_r + i * 32;      // +32 rows per step (256 thr / 8 per row)
            int gm = blockM + r;
            int gk = k0 + a_kq;
            bool okm = gm < M;
            bool okk = gk < K;
            size_t off = (size_t)(okm ? gm : 0) * lda + (okk ? gk : 0);
            float4 v = *(const float4*)(A + off);
            if (i == 0) __builtin_prefetch(A + off + TK, 0, 0);
            if (!(okm && okk)) { v.x = 0.f; v.y = 0.f; v.z = 0.f; v.w = 0.f; }
            uint2 pk;
            pk.x = ((unsigned)f2bf(v.y) << 16) | f2bf(v.x);
            pk.y = ((unsigned)f2bf(v.w) << 16) | f2bf(v.z);
            *(uint2*)&As[r * AST + a_kq] = pk;
        }
        // ---- stage B tile transposed Bs[n][k]: 4x (float4 -> 4 ds_store_b16) ----
        #pragma unroll
        for (int i = 0; i < 4; ++i) {
            int k  = b_k + i * 8;       // +8 k-rows per step (256 thr / 32 per row)
            int gk = k0 + k;
            bool okk = gk < K;
            size_t off = (size_t)(okk ? gk : 0) * ldb + (blockN + b_nq);
            float4 v = *(const float4*)(Bm + off);
            if (!okk) { v.x = 0.f; v.y = 0.f; v.z = 0.f; v.w = 0.f; }
            Bs[(b_nq + 0) * BST + k] = f2bf(v.x);
            Bs[(b_nq + 1) * BST + k] = f2bf(v.y);
            Bs[(b_nq + 2) * BST + k] = f2bf(v.z);
            Bs[(b_nq + 3) * BST + k] = f2bf(v.w);
        }
        __syncthreads();

        // ---- fragments per ISA 16-bit layouts ----
        union Frag { v16bf v; uint4 q[2]; };
        Frag aF[2], bF[4];
        #pragma unroll
        for (int sm = 0; sm < 2; ++sm) {
            // A (16x32): lane row = l16; e0..7 -> K=8*half+0..7, e8..15 -> +16
            const unsigned short* ar = &As[(waveM * 32 + sm * 16 + l16) * AST + 8 * half];
            aF[sm].q[0] = *(const uint4*)(ar);
            aF[sm].q[1] = *(const uint4*)(ar + 16);
        }
        #pragma unroll
        for (int sn = 0; sn < 4; ++sn) {
            // B (32x16): lane col = l16; e0..7 -> K=8*half+0..7, e8..15 -> +16
            const unsigned short* br = &Bs[(waveN * 64 + sn * 16 + l16) * BST + 8 * half];
            bF[sn].q[0] = *(const uint4*)(br);
            bF[sn].q[1] = *(const uint4*)(br + 16);
        }
        #pragma unroll
        for (int sm = 0; sm < 2; ++sm)
            #pragma unroll
            for (int sn = 0; sn < 4; ++sn)
                acc[sm][sn] = __builtin_amdgcn_wmma_f32_16x16x32_bf16(
                    false, aF[sm].v, false, bF[sn].v, (short)0, acc[sm][sn], false, false);
        __syncthreads();
    }

    // ---- store (C/D layout: lanes0-15 N=lane, VGPR i -> M=i; lanes16-31 M=8+i)
    int   ncol[4];
    float bval[4];
    #pragma unroll
    for (int sn = 0; sn < 4; ++sn) {
        ncol[sn] = blockN + waveN * 64 + sn * 16 + l16;
        bval[sn] = bias ? bias[ncol[sn]] : 0.0f;
    }
    #pragma unroll
    for (int sm = 0; sm < 2; ++sm) {
        #pragma unroll
        for (int i = 0; i < 8; ++i) {
            int m = blockM + waveM * 32 + sm * 16 + i + 8 * half;
            if (m < M) {
                #pragma unroll
                for (int sn = 0; sn < 4; ++sn)
                    C[(size_t)m * ldc + ncol[sn]] = acc[sm][sn][i] + bval[sn];
            }
        }
    }
}

// ---------------------------------------------------------------------------
// Hypergraph scatter machinery
// ---------------------------------------------------------------------------
__global__ void k_degree(const int* __restrict__ he, const int* __restrict__ nd,
                         float* __restrict__ de, float* __restrict__ dn, int nnz)
{
    int i = blockIdx.x * blockDim.x + threadIdx.x;
    if (i < nnz) {
        atomicAdd(&de[he[i]], 1.0f);
        atomicAdd(&dn[nd[i]], 1.0f);
    }
}

// dst[didx[nz]][:] += src[sidx[nz]][:]   (HC_ channels, float4 per thread)
__global__ void k_scatter(const float* __restrict__ src, const int* __restrict__ sidx,
                          const int* __restrict__ didx, float* __restrict__ dst, int nnz)
{
    int i = blockIdx.x * blockDim.x + threadIdx.x;   // nnz * 64 threads
    int nz = i >> 6, c4 = (i & 63) << 2;
    if (nz >= nnz) return;
    const float4 v = *(const float4*)(src + (size_t)sidx[nz] * HC_ + c4);
    float* d = dst + (size_t)didx[nz] * HC_ + c4;
    atomicAdd(d + 0, v.x); atomicAdd(d + 1, v.y);
    atomicAdd(d + 2, v.z); atomicAdd(d + 3, v.w);
}

__global__ void k_scale_rows(float* __restrict__ m, const float* __restrict__ deg, int rows)
{
    int i = blockIdx.x * blockDim.x + threadIdx.x;   // rows * 64
    int r = i >> 6, c4 = (i & 63) << 2;
    if (r >= rows) return;
    float d = deg[r];
    float s = d > 0.0f ? 1.0f / d : 0.0f;
    float4* p = (float4*)(m + (size_t)r * HC_ + c4);
    float4 v = *p;
    v.x *= s; v.y *= s; v.z *= s; v.w *= s;
    *p = v;
}

// acc = tanh(acc * Dinv + bias); mean-pool into hpool (400 nodes/graph)
__global__ void k_finish(float* __restrict__ acc, const float* __restrict__ dn,
                         const float* __restrict__ bias, float* __restrict__ hpool, int loff)
{
    int i = blockIdx.x * blockDim.x + threadIdx.x;   // N_*HC_
    if (i >= N_ * HC_) return;
    int n = i >> 8, c = i & 255;
    float d = dn[n];
    float s = d > 0.0f ? 1.0f / d : 0.0f;
    float v = tanhf(acc[i] * s + bias[c]);
    acc[i] = v;
    atomicAdd(&hpool[(n / R_) * (HC_ * L_) + loff + c], v * (1.0f / R_));
}

// ---------------------------------------------------------------------------
// BatchNorm over batch axis (64 rows): one thread per column (coalesced rows)
// ---------------------------------------------------------------------------
__global__ void k_bn_cols(const float* __restrict__ X, int ldx,
                          const float* __restrict__ g, const float* __restrict__ bt,
                          float* __restrict__ Y, int ldy, int C, int relu)
{
    int c = blockIdx.x * blockDim.x + threadIdx.x;
    if (c >= C) return;
    float s = 0.0f, s2 = 0.0f;
    for (int r = 0; r < B_; ++r) {
        float v = X[(size_t)r * ldx + c];
        s += v; s2 += v * v;
    }
    float mu  = s * (1.0f / B_);
    float var = s2 * (1.0f / B_) - mu * mu;
    float inv = rsqrtf(var + EPS_) * g[c];
    float bb  = bt[c];
    for (int r = 0; r < B_; ++r) {
        float v = (X[(size_t)r * ldx + c] - mu) * inv + bb;
        if (relu) v = fmaxf(v, 0.0f);
        Y[(size_t)r * ldy + c] = v;
    }
}

// upper-tri extraction of corr + BN, written straight into z[:, 0:CV_]
__global__ void k_bn_corr(const float* __restrict__ corr, const float* __restrict__ g,
                          const float* __restrict__ bt, float* __restrict__ z)
{
    int v = blockIdx.x * blockDim.x + threadIdx.x;
    if (v >= CV_) return;
    int i = (int)((2.0 * R_ + 1.0 -
                   sqrt((2.0 * R_ + 1.0) * (2.0 * R_ + 1.0) - 8.0 * (double)v)) * 0.5);
    if (i < 0) i = 0;
    if (i > R_ - 1) i = R_ - 1;
    while (i > 0 && triu_off(i) > v) --i;
    while (i < R_ - 1 && triu_off(i + 1) <= v) ++i;
    int j = i + (v - triu_off(i));
    const float* base = corr + (size_t)i * R_ + j;
    float s = 0.0f, s2 = 0.0f;
    for (int b = 0; b < B_; ++b) {
        float x = base[(size_t)b * R_ * R_];
        s += x; s2 += x * x;
    }
    float mu  = s * (1.0f / B_);
    float var = s2 * (1.0f / B_) - mu * mu;
    float inv = rsqrtf(var + EPS_) * g[v];
    float bb  = bt[v];
    for (int b = 0; b < B_; ++b)
        z[(size_t)b * IN1_ + v] = (base[(size_t)b * R_ * R_] - mu) * inv + bb;
}

// ---------------------------------------------------------------------------
// final [64,256] x [256,2] + bias + log_softmax.
// Wm4 (2 KB) is broadcast to every row -> stage it into LDS with the Tensor
// Data Mover (1D tile: 512 f32, tile==tensor dims), then dot out of LDS.
// ---------------------------------------------------------------------------
__global__ __launch_bounds__(64)
void k_head(const float* __restrict__ a3, const float* __restrict__ W,
            const float* __restrict__ bv, float* __restrict__ out)
{
    __shared__ __align__(16) float Wl[(HID_ / 2) * NC_];   // 512 f32 @ LDS offset 0

#if defined(__gfx1250__) && __has_builtin(__builtin_amdgcn_tensor_load_to_lds)
    if (threadIdx.x == 0) {   // TDM ignores EXEC; issue from one wave only
        typedef __attribute__((ext_vector_type(4))) unsigned int v4u;
        typedef __attribute__((ext_vector_type(8))) int v8i;
        typedef __attribute__((ext_vector_type(4))) int v4i;
        unsigned long long ga = (unsigned long long)(uintptr_t)W;
        v4u g0;
        g0[0] = 1u;                                   // count=1, user descriptor
        g0[1] = 0u;                                   // lds_addr = 0 (Wl at base)
        g0[2] = (unsigned int)ga;                     // global_addr[31:0]
        g0[3] = (unsigned int)((ga >> 32) & 0x01FFFFFFu) | (2u << 30); // addr[56:32] | type=2
        v8i g1;
        g1[0] = (int)(2u << 16);                      // data_size=2 (4 bytes)
        g1[1] = (int)(512u << 16);                    // tensor_dim0[15:0]=512
        g1[2] = (int)(1u << 16);                      // tensor_dim1[15:0]=1
        g1[3] = (int)(512u << 16);                    // tile_dim0=512
        g1[4] = 1;                                    // tile_dim1=1
        g1[5] = 512;                                  // tensor_dim0_stride=512
        g1[6] = 0; g1[7] = 0;
        v4i gz4 = {0, 0, 0, 0};
        v8i gz8 = {0, 0, 0, 0, 0, 0, 0, 0};
        // 6-arg form (clang-23 / therock-10.0 headers): (g0, g1, g2, g3, g4, cpol)
        __builtin_amdgcn_tensor_load_to_lds(g0, g1, gz4, gz4, gz8, 0);
        __builtin_amdgcn_s_wait_tensorcnt(0);
    }
    __syncthreads();
#else
    for (int i = threadIdx.x; i < (HID_ / 2) * NC_; i += blockDim.x) Wl[i] = W[i];
    __syncthreads();
#endif

    int r = threadIdx.x;
    if (r >= B_) return;
    float l0 = bv[0], l1 = bv[1];
    const float* x = a3 + (size_t)r * (HID_ / 2);
    for (int k = 0; k < HID_ / 2; ++k) {
        float xv = x[k];
        l0 += xv * Wl[k * NC_ + 0];
        l1 += xv * Wl[k * NC_ + 1];
    }
    float m   = fmaxf(l0, l1);
    float lse = m + logf(expf(l0 - m) + expf(l1 - m));
    out[r * NC_ + 0] = l0 - lse;
    out[r * NC_ + 1] = l1 - lse;
}

// ---------------------------------------------------------------------------
extern "C" void kernel_launch(void* const* d_in, const int* in_sizes, int n_in,
                              void* d_out, int out_size, void* d_ws, size_t ws_size,
                              hipStream_t stream)
{
    (void)in_sizes; (void)n_in; (void)out_size; (void)ws_size;
    const float* x        = (const float*)d_in[0];
    const float* corr     = (const float*)d_in[1];
    const int*   node_idx = (const int*)d_in[2];
    const int*   he_idx   = (const int*)d_in[3];
    /* d_in[4] = batch, implied by n / R_ */
    const float* W1 = (const float*)d_in[5];   const float* b1  = (const float*)d_in[6];
    const float* W2 = (const float*)d_in[7];   const float* b2  = (const float*)d_in[8];
    const float* W3 = (const float*)d_in[9];   const float* b3  = (const float*)d_in[10];
    const float* bnh_g = (const float*)d_in[11]; const float* bnh_b = (const float*)d_in[12];
    const float* bn_g  = (const float*)d_in[13]; const float* bn_b  = (const float*)d_in[14];
    const float* Wm1 = (const float*)d_in[15]; const float* bm1 = (const float*)d_in[16];
    const float* g1  = (const float*)d_in[17]; const float* be1 = (const float*)d_in[18];
    const float* Wm2 = (const float*)d_in[19]; const float* bm2 = (const float*)d_in[20];
    const float* g2  = (const float*)d_in[21]; const float* be2 = (const float*)d_in[22];
    const float* Wm3 = (const float*)d_in[23]; const float* bm3 = (const float*)d_in[24];
    const float* g3  = (const float*)d_in[25]; const float* be3 = (const float*)d_in[26];
    const float* Wm4 = (const float*)d_in[27]; const float* bm4 = (const float*)d_in[28];

    float* ws = (float*)d_ws;
    size_t off = 0;
    float* xt    = ws + off; off += (size_t)N_ * HC_;
    float* mbuf  = ws + off; off += (size_t)E_ * HC_;
    float* acc   = ws + off; off += (size_t)N_ * HC_;
    float* de    = ws + off; off += E_;
    float* dn    = ws + off; off += N_;
    float* hpool = ws + off; off += (size_t)B_ * HC_ * L_;
    float* z     = ws + off; off += (size_t)B_ * IN1_;
    float* a1    = ws + off; off += (size_t)B_ * HID_;
    float* a2    = ws + off; off += (size_t)B_ * (HID_ / 2);
    float* a3    = ws + off; off += (size_t)B_ * (HID_ / 2);

    (void)hipMemsetAsync(de, 0, E_ * sizeof(float), stream);
    (void)hipMemsetAsync(dn, 0, N_ * sizeof(float), stream);
    (void)hipMemsetAsync(hpool, 0, (size_t)B_ * HC_ * L_ * sizeof(float), stream);
    k_degree<<<(NNZ_ + 255) / 256, 256, 0, stream>>>(he_idx, node_idx, de, dn, NNZ_);

    const float* curA = x;
    int curK = R_;
    const float* Ws[3] = {W1, W2, W3};
    const float* bs[3] = {b1, b2, b3};
    const int scat_blocks = (NNZ_ * 64) / 256;
    for (int l = 0; l < 3; ++l) {
        dim3 gg(HC_ / TN, (N_ + TM - 1) / TM);
        k_gemm<<<gg, 256, 0, stream>>>(curA, Ws[l], nullptr, xt, N_, HC_, curK, curK, HC_, HC_);
        (void)hipMemsetAsync(mbuf, 0, (size_t)E_ * HC_ * sizeof(float), stream);
        k_scatter<<<scat_blocks, 256, 0, stream>>>(xt, node_idx, he_idx, mbuf, NNZ_);
        k_scale_rows<<<(E_ * 64) / 256, 256, 0, stream>>>(mbuf, de, E_);
        (void)hipMemsetAsync(acc, 0, (size_t)N_ * HC_ * sizeof(float), stream);
        k_scatter<<<scat_blocks, 256, 0, stream>>>(mbuf, he_idx, node_idx, acc, NNZ_);
        k_finish<<<(N_ * HC_ + 255) / 256, 256, 0, stream>>>(acc, dn, bs[l], hpool, l * HC_);
        curA = acc;
        curK = HC_;
    }

    k_bn_corr<<<(CV_ + 255) / 256, 256, 0, stream>>>(corr, bn_g, bn_b, z);
    k_bn_cols<<<(HC_ * L_ + 255) / 256, 256, 0, stream>>>(hpool, HC_ * L_, bnh_g, bnh_b,
                                                          z + CV_, IN1_, HC_ * L_, 0);
    {
        dim3 gg(HID_ / TN, 1);
        k_gemm<<<gg, 256, 0, stream>>>(z, Wm1, bm1, a1, B_, HID_, IN1_, IN1_, HID_, HID_);
    }
    k_bn_cols<<<(HID_ + 255) / 256, 256, 0, stream>>>(a1, HID_, g1, be1, a1, HID_, HID_, 1);
    {
        dim3 gg((HID_ / 2) / TN, 1);
        k_gemm<<<gg, 256, 0, stream>>>(a1, Wm2, bm2, a2, B_, HID_ / 2, HID_, HID_, HID_ / 2, HID_ / 2);
    }
    k_bn_cols<<<((HID_ / 2) + 255) / 256, 256, 0, stream>>>(a2, HID_ / 2, g2, be2, a2, HID_ / 2, HID_ / 2, 1);
    {
        dim3 gg((HID_ / 2) / TN, 1);
        k_gemm<<<gg, 256, 0, stream>>>(a2, Wm3, bm3, a3, B_, HID_ / 2, HID_ / 2, HID_ / 2, HID_ / 2, HID_ / 2);
    }
    k_bn_cols<<<((HID_ / 2) + 255) / 256, 256, 0, stream>>>(a3, HID_ / 2, g3, be3, a3, HID_ / 2, HID_ / 2, 1);
    k_head<<<1, 64, 0, stream>>>(a3, Wm4, bm4, (float*)d_out);
}
